// EntityRelationJointEnhancer_71227737637025
// MI455X (gfx1250) — matrix operand order repaired and checked by reading, not compiled, not run on hardware
//
#include <hip/hip_runtime.h>
#include <hip/hip_fp16.h>

typedef __attribute__((ext_vector_type(16))) _Float16 v16h;
typedef __attribute__((ext_vector_type(8)))  _Float16 v8h;
typedef __attribute__((ext_vector_type(8)))  float    v8f;

__device__ __forceinline__ v8f wmma32f16(v16h a, v16h b, v8f c) {
    return __builtin_amdgcn_wmma_f32_16x16x32_f16(
        /*neg_a=*/false, a, /*neg_b=*/false, b,
        /*c_mod=*/(short)0, c, /*reuse_a=*/false, /*reuse_b=*/false);
}

// ---------------- K0: zero accumulators ----------------
__global__ void k_zero(float* __restrict__ p, long n) {
    long i = (long)blockIdx.x * blockDim.x + threadIdx.x;
    if (i < n) p[i] = 0.0f;
}

// ---------------- K1: rel_ctx mean + f16 weight conversion ----------------
__global__ void k_prep(const float* __restrict__ rel, int R,
                       const float* __restrict__ w1a, const float* __restrict__ w2a,
                       const float* __restrict__ w1b, const float* __restrict__ w2b,
                       float* __restrict__ ctx, _Float16* __restrict__ ctx_h,
                       _Float16* __restrict__ w1a_h, _Float16* __restrict__ w2a_h,
                       _Float16* __restrict__ w1b_h, _Float16* __restrict__ w2b_h) {
    int t = threadIdx.x;   // 256 threads, one block
    if (t < 64) {
        float s = 0.0f;
        for (int r = 0; r < R; ++r) s += rel[r * 64 + t];
        float m = s / (float)R;
        ctx[t] = m;
        ctx_h[t] = (_Float16)m;
    }
    for (int i = t; i < 64 * 128; i += 256) {
        w1a_h[i] = (_Float16)w1a[i];
        w1b_h[i] = (_Float16)w1b[i];
    }
    for (int i = t; i < 64 * 64; i += 256) {
        w2a_h[i] = (_Float16)w2a[i];
        w2b_h[i] = (_Float16)w2b[i];
    }
}

// ---------------- K2: edge scatter (4 threads per edge, 16 floats each) ----------------
__global__ void k_edges(const long long* __restrict__ ei, const long long* __restrict__ et,
                        const float* __restrict__ rel,
                        float* __restrict__ sum_feat, float* __restrict__ deg,
                        float* __restrict__ nbr, long E) {
    long tid = (long)blockIdx.x * blockDim.x + threadIdx.x;
    long e = tid >> 2;
    if (e >= E) return;
    int c = (int)(tid & 3);                 // 16-float chunk
    long src = ei[e];
    long dst = ei[E + e];
    long rt  = et[e];
    bool ns = (src != dst);
    const float4* rp = (const float4*)(rel + rt * 64 + c * 16);
    float4 v0 = rp[0], v1 = rp[1], v2 = rp[2], v3 = rp[3];
    float vals[16] = { v0.x, v0.y, v0.z, v0.w, v1.x, v1.y, v1.z, v1.w,
                       v2.x, v2.y, v2.z, v2.w, v3.x, v3.y, v3.z, v3.w };
    float* ps = sum_feat + src * 64 + c * 16;
    #pragma unroll
    for (int i = 0; i < 16; ++i) atomicAdd(ps + i, vals[i]);
    if (ns) {
        float* pd = sum_feat + dst * 64 + c * 16;
        #pragma unroll
        for (int i = 0; i < 16; ++i) atomicAdd(pd + i, vals[i]);
    }
    if (c == 0) {
        atomicAdd(deg + src, 1.0f);
        if (ns) {
            atomicAdd(deg + dst, 1.0f);
            atomicAdd(nbr + src, 1.0f);
            atomicAdd(nbr + dst, 1.0f);
        }
    }
}

// ---------------- K3: feat = sum/deg (or ctx fallback), f32 in-place + f16 copy ------
__global__ void k_feat(const float* __restrict__ ctx, const float* __restrict__ deg,
                       float* __restrict__ sum_feat, _Float16* __restrict__ feat_h,
                       long n64) {
    long i = (long)blockIdx.x * blockDim.x + threadIdx.x;
    if (i >= n64) return;
    long n = i >> 6;
    int  d = (int)(i & 63);
    float dg = deg[n];
    float f = (dg > 0.0f) ? sum_feat[i] / fmaxf(dg, 1.0f) : ctx[d];
    sum_feat[i] = f;
    feat_h[i] = (_Float16)f;
}

// ---------------- K4: dual-MLP via WMMA + final blend ----------------
__launch_bounds__(256)
__global__ void k_mlp(const float* __restrict__ feat, const _Float16* __restrict__ feat_h,
                      const float* __restrict__ ctx, const _Float16* __restrict__ ctx_h,
                      const _Float16* __restrict__ w1a_h, const float* __restrict__ b1a,
                      const _Float16* __restrict__ w2a_h, const float* __restrict__ b2a,
                      const _Float16* __restrict__ w1b_h, const float* __restrict__ b1b,
                      const _Float16* __restrict__ w2b_h, const float* __restrict__ b2b,
                      const float* __restrict__ deg, const float* __restrict__ nbr,
                      const float* __restrict__ strength,
                      float* __restrict__ out, int ntiles, long N) {
    __shared__ __align__(32) _Float16 hbuf[8][16 * 64];  // per-wave layer-1 output staging

    int lane = threadIdx.x & 31;
    int w    = threadIdx.x >> 5;
    int tile = blockIdx.x * 8 + w;
    if (tile >= ntiles) return;
    long base = (long)tile * 16;
    int row = lane & 15;          // A: row m ; B: column o (within 16-tile)
    int hi  = (lane >> 4) & 1;    // upper half-wave

    // A-fragment: 16x32 f16 tile from a 64-wide source row; kt selects 32-chunk
    auto loadA = [&](const _Float16* srcRow, int kt) -> v16h {
        int klo = kt * 32 + (hi ? 16 : 0);
        v8h lo = *(const v8h*)(srcRow + klo);
        v8h hh = *(const v8h*)(srcRow + klo + 8);
        v16h a;
        #pragma unroll
        for (int j = 0; j < 8; ++j) { a[j] = lo[j]; a[8 + j] = hh[j]; }
        return a;
    };
    // B-fragment: 32x16 f16 tile; wrow = weight row for this lane's output channel
    auto loadB = [&](const _Float16* wrow, int kOff) -> v16h {
        return *(const v16h*)(wrow + kOff + (hi ? 16 : 0));
    };

    long nodeA = base + row;                 // A-operand node row (guarded)
    if (nodeA >= N) nodeA = N - 1;
    const _Float16* myFeat = feat_h + nodeA * 64;

    v16h a0  = loadA(myFeat, 0), a1  = loadA(myFeat, 1);   // feat half of x
    v16h ac0 = loadA(ctx_h, 0),  ac1 = loadA(ctx_h, 1);    // ctx half of x (MLP a)

    v8f ya[4], yb[4];

    // ================= MLP a =================
    #pragma unroll
    for (int nt = 0; nt < 4; ++nt) {
        int o = nt * 16 + row;
        float bb = b1a[o];
        v8f c = { bb, bb, bb, bb, bb, bb, bb, bb };
        const _Float16* wrow = w1a_h + o * 128;
        c = wmma32f16(a0,  loadB(wrow, 0),  c);
        c = wmma32f16(a1,  loadB(wrow, 32), c);
        c = wmma32f16(ac0, loadB(wrow, 64), c);
        c = wmma32f16(ac1, loadB(wrow, 96), c);
        #pragma unroll
        for (int r = 0; r < 8; ++r)
            hbuf[w][(r + 8 * hi) * 64 + o] = (_Float16)fmaxf(c[r], 0.0f);
    }
    {
        const _Float16* hrow = &hbuf[w][row * 64];
        v16h h0 = loadA(hrow, 0), h1 = loadA(hrow, 1);
        #pragma unroll
        for (int nt = 0; nt < 4; ++nt) {
            int o = nt * 16 + row;
            float bb = b2a[o];
            v8f c = { bb, bb, bb, bb, bb, bb, bb, bb };
            const _Float16* wrow = w2a_h + o * 64;
            c = wmma32f16(h0, loadB(wrow, 0),  c);
            c = wmma32f16(h1, loadB(wrow, 32), c);
            ya[nt] = c;
        }
    }

    // ================= MLP b  (x = [feat | feat]) =================
    #pragma unroll
    for (int nt = 0; nt < 4; ++nt) {
        int o = nt * 16 + row;
        float bb = b1b[o];
        v8f c = { bb, bb, bb, bb, bb, bb, bb, bb };
        const _Float16* wrow = w1b_h + o * 128;
        c = wmma32f16(a0, loadB(wrow, 0),  c);
        c = wmma32f16(a1, loadB(wrow, 32), c);
        c = wmma32f16(a0, loadB(wrow, 64), c);
        c = wmma32f16(a1, loadB(wrow, 96), c);
        #pragma unroll
        for (int r = 0; r < 8; ++r)
            hbuf[w][(r + 8 * hi) * 64 + o] = (_Float16)fmaxf(c[r], 0.0f);
    }
    {
        const _Float16* hrow = &hbuf[w][row * 64];
        v16h h0 = loadA(hrow, 0), h1 = loadA(hrow, 1);
        #pragma unroll
        for (int nt = 0; nt < 4; ++nt) {
            int o = nt * 16 + row;
            float bb = b2b[o];
            v8f c = { bb, bb, bb, bb, bb, bb, bb, bb };
            const _Float16* wrow = w2b_h + o * 64;
            c = wmma32f16(h0, loadB(wrow, 0),  c);
            c = wmma32f16(h1, loadB(wrow, 32), c);
            yb[nt] = c;
        }
    }

    // ================= final blend =================
    float s = strength[0];
    s = fminf(fmaxf(s, 0.0f), 0.3f);
    #pragma unroll
    for (int nt = 0; nt < 4; ++nt) {
        int o = nt * 16 + row;
        float cx = ctx[o];
        #pragma unroll
        for (int r = 0; r < 8; ++r) {
            long node = base + r + 8 * hi;
            if (node >= N) continue;
            float dg = deg[node];
            float nb = nbr[node];
            float f  = feat[node * 64 + o];
            float inter = ya[nt][r];
            float cf = (nb > 0.0f) ? yb[nt][r] : inter;
            float enh = (1.0f - s) * f + s * cf;
            out[node * 64 + o] = (dg > 0.0f) ? enh : cx;
        }
    }
}

extern "C" void kernel_launch(void* const* d_in, const int* in_sizes, int n_in,
                              void* d_out, int out_size, void* d_ws, size_t ws_size,
                              hipStream_t stream) {
    const long long* ei  = (const long long*)d_in[0];   // edge_index [2,E] int64
    const long long* et  = (const long long*)d_in[1];   // edge_type  [E]   int64
    const float* rel     = (const float*)d_in[2];       // [R,64]
    const float* w1a     = (const float*)d_in[3];
    const float* b1a     = (const float*)d_in[4];
    const float* w2a     = (const float*)d_in[5];
    const float* b2a     = (const float*)d_in[6];
    const float* w1b     = (const float*)d_in[7];
    const float* b1b     = (const float*)d_in[8];
    const float* w2b     = (const float*)d_in[9];
    const float* b2b     = (const float*)d_in[10];
    const float* strength= (const float*)d_in[11];
    float* out = (float*)d_out;

    long E = in_sizes[1];
    int  R = in_sizes[2] / 64;
    long N = (long)out_size / 64;

    // workspace carve (no padding: segments 32B-aligned only if N % 8 == 0)
    float* sum_feat = (float*)d_ws;                 // N*64 f32 (becomes feat)
    float* deg      = sum_feat + N * 64;            // N
    float* nbr      = deg + N;                      // N
    float* ctx      = nbr + N;                      // 64
    _Float16* ctx_h = (_Float16*)(ctx + 64);        // 64
    _Float16* feat_h= ctx_h + 64;                   // N*64
    _Float16* w1a_h = feat_h + N * 64;              // 64*128
    _Float16* w2a_h = w1a_h + 64 * 128;             // 64*64
    _Float16* w1b_h = w2a_h + 64 * 64;              // 64*128
    _Float16* w2b_h = w1b_h + 64 * 128;             // 64*64

    long nz = N * 64 + 2 * N;
    k_zero<<<(int)((nz + 255) / 256), 256, 0, stream>>>(sum_feat, nz);

    k_prep<<<1, 256, 0, stream>>>(rel, R, w1a, w2a, w1b, w2b,
                                  ctx, ctx_h, w1a_h, w2a_h, w1b_h, w2b_h);

    long nedge_threads = E * 4;
    k_edges<<<(int)((nedge_threads + 255) / 256), 256, 0, stream>>>(
        ei, et, rel, sum_feat, deg, nbr, E);

    long n64 = N * 64;
    k_feat<<<(int)((n64 + 255) / 256), 256, 0, stream>>>(ctx, deg, sum_feat, feat_h, n64);

    int ntiles = (int)((N + 15) / 16);
    k_mlp<<<(ntiles + 7) / 8, 256, 0, stream>>>(
        sum_feat, feat_h, ctx, ctx_h,
        w1a_h, b1a, w2a_h, b2a, w1b_h, b1b, w2b_h, b2b,
        deg, nbr, strength, out, ntiles, N);
}
